// MyLayer_33543694581930
// MI455X (gfx1250) — matrix operand order repaired and checked by reading, not compiled, not run on hardware
//
#include <hip/hip_runtime.h>

typedef _Float16 f16;
typedef __attribute__((ext_vector_type(16))) _Float16 v16h;
typedef __attribute__((ext_vector_type(8)))  _Float16 v8h;
typedef __attribute__((ext_vector_type(8)))  float    v8f;
typedef __attribute__((ext_vector_type(4)))  float    v4f;

#define BATCH 16384
#define UNITS 1024
#define HID   128
#define DT    0.01f
#define NSTEP 20

// ---------------------------------------------------------------------------
// Prep: W1[k][n] (1024x128) -> W1t[n][k] f16 ;  W2[k][n] (128x1024) -> W2t[n][k] f16
// Both have 131072 elements.
// ---------------------------------------------------------------------------
__global__ __launch_bounds__(256)
void ode_prep_weights_k(const float* __restrict__ W1, const float* __restrict__ W2,
                        f16* __restrict__ W1t, f16* __restrict__ W2t)
{
  int i = blockIdx.x * 256 + threadIdx.x;            // 0 .. 131071
  {
    int n = i >> 10, k = i & 1023;                   // W1t is 128 x 1024
    W1t[i] = (f16)W1[(size_t)k * HID + n];
  }
  {
    int n = i >> 7, k = i & 127;                     // W2t is 1024 x 128
    W2t[i] = (f16)W2[(size_t)k * UNITS + n];
  }
}

// ---------------------------------------------------------------------------
// GEMM1: h = relu(vin[16384x1024] @ W1 + b1) -> h f16 [16384x128]
// Block = 128 threads (4 waves). Wave w computes rows [m0, m0+16) x all 128 cols
// as 8 accumulators of v_wmma_f32_16x16x32_f16.
// A layout (16-bit, 16x32): lane<16: M=lane, K elems {0..7,16..23};
//                           lane>=16: M=lane-16, K elems {8..15,24..31}.
// B layout (16-bit, 32x16): lane<16: N=lane, K {0..7,16..23}; lane>=16: N=lane-16, K {8..15,24..31}.
// All 8 B fragments are loaded into distinct registers before the 8 WMMAs so the
// loads clause together and the WMMAs issue back-to-back.
// ---------------------------------------------------------------------------
__global__ __launch_bounds__(128)
void ode_gemm1_relu_k(const float* __restrict__ vin, const f16* __restrict__ W1t,
                      const float* __restrict__ b1, f16* __restrict__ h)
{
  const int lane = threadIdx.x & 31;
  const int wave = threadIdx.x >> 5;
  const int half = lane >> 4;
  const int l15  = lane & 15;
  const int m0   = blockIdx.x * 64 + wave * 16;
  const int row  = m0 + l15;

  v8f acc[8];
#pragma unroll
  for (int i = 0; i < 8; ++i)
#pragma unroll
    for (int j = 0; j < 8; ++j) acc[i][j] = 0.0f;

  const float* arow = vin + (size_t)row * UNITS + half * 8;
  const f16*   brow = W1t + (size_t)l15 * UNITS + half * 8;

  for (int kc = 0; kc < UNITS; kc += 32) {
    // A fragment: two contiguous 8-float runs per lane, converted f32->f16
    const float* ap = arow + kc;
    v4f a0 = *(const v4f*)(ap);
    v4f a1 = *(const v4f*)(ap + 4);
    v4f a2 = *(const v4f*)(ap + 16);
    v4f a3 = *(const v4f*)(ap + 20);
    v16h av;
#pragma unroll
    for (int i = 0; i < 4; ++i) {
      av[i]      = (f16)a0[i];
      av[4 + i]  = (f16)a1[i];
      av[8 + i]  = (f16)a2[i];
      av[12 + i] = (f16)a3[i];
    }
    // All 8 B fragments up front (independent registers -> clause + deep overlap)
    v16h bv[8];
#pragma unroll
    for (int nt = 0; nt < 8; ++nt) {
      const f16* bp = brow + kc + (size_t)nt * 16 * UNITS;
      v8h b0 = *(const v8h*)(bp);
      v8h b1v = *(const v8h*)(bp + 16);
#pragma unroll
      for (int i = 0; i < 8; ++i) { bv[nt][i] = b0[i]; bv[nt][8 + i] = b1v[i]; }
    }
#pragma unroll
    for (int nt = 0; nt < 8; ++nt) {
      acc[nt] = __builtin_amdgcn_wmma_f32_16x16x32_f16(false, av, false, bv[nt],
                                                       (short)0, acc[nt], false, false);
    }
  }

  // Epilogue: C/D layout — VGPR r: row = m0 + r + 8*half, col = lane&15 (+16*nt)
#pragma unroll
  for (int nt = 0; nt < 8; ++nt) {
    int col = nt * 16 + l15;
    float bias = b1[col];
#pragma unroll
    for (int r = 0; r < 8; ++r) {
      float val = acc[nt][r] + bias;
      val = val > 0.0f ? val : 0.0f;
      h[(size_t)(m0 + r + half * 8) * HID + col] = (f16)val;
    }
  }
}

// ---------------------------------------------------------------------------
// GEMM2: z = relu(h[16384x128] @ W2 + b2) -> z f32 [16384x1024]
// Grid (256, 8): blockIdx.x over M/64, blockIdx.y over N/128. K = 128.
// ---------------------------------------------------------------------------
__global__ __launch_bounds__(128)
void ode_gemm2_relu_k(const f16* __restrict__ h, const f16* __restrict__ W2t,
                      const float* __restrict__ b2, float* __restrict__ z)
{
  const int lane = threadIdx.x & 31;
  const int wave = threadIdx.x >> 5;
  const int half = lane >> 4;
  const int l15  = lane & 15;
  const int m0   = blockIdx.x * 64 + wave * 16;
  const int n0   = blockIdx.y * 128;
  const int row  = m0 + l15;

  v8f acc[8];
#pragma unroll
  for (int i = 0; i < 8; ++i)
#pragma unroll
    for (int j = 0; j < 8; ++j) acc[i][j] = 0.0f;

  const f16* arow = h + (size_t)row * HID + half * 8;
  const f16* brow = W2t + (size_t)(n0 + l15) * HID + half * 8;

#pragma unroll
  for (int kc = 0; kc < HID; kc += 32) {
    const f16* ap = arow + kc;
    v8h a0 = *(const v8h*)(ap);
    v8h a1 = *(const v8h*)(ap + 16);
    v16h av;
#pragma unroll
    for (int i = 0; i < 8; ++i) { av[i] = a0[i]; av[8 + i] = a1[i]; }

    v16h bv[8];
#pragma unroll
    for (int nt = 0; nt < 8; ++nt) {
      const f16* bp = brow + kc + (size_t)nt * 16 * HID;
      v8h b0 = *(const v8h*)(bp);
      v8h b1v = *(const v8h*)(bp + 16);
#pragma unroll
      for (int i = 0; i < 8; ++i) { bv[nt][i] = b0[i]; bv[nt][8 + i] = b1v[i]; }
    }
#pragma unroll
    for (int nt = 0; nt < 8; ++nt) {
      acc[nt] = __builtin_amdgcn_wmma_f32_16x16x32_f16(false, av, false, bv[nt],
                                                       (short)0, acc[nt], false, false);
    }
  }

#pragma unroll
  for (int nt = 0; nt < 8; ++nt) {
    int col = n0 + nt * 16 + l15;
    float bias = b2[col];
#pragma unroll
    for (int r = 0; r < 8; ++r) {
      float val = acc[nt][r] + bias;
      val = val > 0.0f ? val : 0.0f;
      z[(size_t)(m0 + r + half * 8) * UNITS + col] = val;
    }
  }
}

// ---------------------------------------------------------------------------
// Row stats for normalize-VJP: per row of vin: max/argmax, min/argmin,
// S = sum(z), T = sum(z*v) - min*S.  One wave32 per row, butterfly reduce.
// stats layout per row (stride 8 f32): [invr, S, T, imin(int), imax(int), ...]
// ---------------------------------------------------------------------------
__global__ __launch_bounds__(256)
void ode_rowstats_k(const float* __restrict__ vin, const float* __restrict__ z,
                    float* __restrict__ stats)
{
  const int lane = threadIdx.x & 31;
  const int wv   = threadIdx.x >> 5;
  const int row  = blockIdx.x * 8 + wv;
  const float* vr = vin + (size_t)row * UNITS;
  const float* zr = z   + (size_t)row * UNITS;

  float vmax = -3.0e38f, vmin = 3.0e38f;
  int imax = 0, imin = 0;
  float sz = 0.0f, szv = 0.0f;
  for (int c = lane; c < UNITS; c += 32) {
    float vv = vr[c];
    float zz = zr[c];
    if (vv > vmax) { vmax = vv; imax = c; }
    if (vv < vmin) { vmin = vv; imin = c; }
    sz  += zz;
    szv += zz * vv;
  }
#pragma unroll
  for (int off = 16; off > 0; off >>= 1) {
    float ovmax = __shfl_xor(vmax, off);
    int   oimax = __shfl_xor(imax, off);
    float ovmin = __shfl_xor(vmin, off);
    int   oimin = __shfl_xor(imin, off);
    sz  += __shfl_xor(sz,  off);
    szv += __shfl_xor(szv, off);
    if (ovmax > vmax || (ovmax == vmax && oimax < imax)) { vmax = ovmax; imax = oimax; }
    if (ovmin < vmin || (ovmin == vmin && oimin < imin)) { vmin = ovmin; imin = oimin; }
  }
  if (lane == 0) {
    float r    = vmax - vmin + 1e-8f;
    float invr = 1.0f / r;
    float* s   = stats + (size_t)row * 8;
    s[0] = invr;
    s[1] = sz;
    s[2] = szv - vmin * sz;   // T
    int* si = (int*)s;
    si[3] = imin;
    si[4] = imax;
  }
}

// ---------------------------------------------------------------------------
// Apply VJP + RK4 update:
//   g = z*invr;  g[imin] += (T/r - S)/r;  g[imax] -= T/r^2
//   y    += wk * g
//   vnext = v + ck * g        (only when writeV)
// ---------------------------------------------------------------------------
__global__ __launch_bounds__(256)
void ode_apply_update_k(const float* __restrict__ z, const float* __restrict__ stats,
                        const float* __restrict__ v, float* __restrict__ y,
                        float* __restrict__ vnext, float wk, float ck, int writeV)
{
  int tid = blockIdx.x * 256 + threadIdx.x;   // BATCH*UNITS/4 threads
  int row = tid >> 8;                         // 256 float4 per row
  int c0  = (tid & 255) << 2;
  const float* s = stats + (size_t)row * 8;
  float invr = s[0], S = s[1], T = s[2];
  const int* si = (const int*)s;
  int imin = si[3], imax = si[4];
  float corrMin = (T * invr - S) * invr;
  float corrMax = -(T * invr) * invr;

  size_t base = (size_t)row * UNITS + c0;
  v4f zv = *(const v4f*)(z + base);
  v4f g;
#pragma unroll
  for (int i = 0; i < 4; ++i) {
    int c = c0 + i;
    float gg = zv[i] * invr;
    if (c == imin) gg += corrMin;
    if (c == imax) gg += corrMax;
    g[i] = gg;
  }
  v4f yv = *(const v4f*)(y + base);
#pragma unroll
  for (int i = 0; i < 4; ++i) yv[i] += wk * g[i];
  *(v4f*)(y + base) = yv;
  if (writeV) {
    v4f vv = *(const v4f*)(v + base);
#pragma unroll
    for (int i = 0; i < 4; ++i) vv[i] += ck * g[i];
    *(v4f*)(vnext + base) = vv;
  }
}

// ---------------------------------------------------------------------------
extern "C" void kernel_launch(void* const* d_in, const int* in_sizes, int n_in,
                              void* d_out, int out_size, void* d_ws, size_t ws_size,
                              hipStream_t stream)
{
  (void)in_sizes; (void)n_in; (void)out_size; (void)ws_size;

  const float* x  = (const float*)d_in[0];
  const float* W1 = (const float*)d_in[1];
  const float* b1 = (const float*)d_in[2];
  const float* W2 = (const float*)d_in[3];
  const float* b2 = (const float*)d_in[4];
  float* v = (float*)d_out;                         // current ODE state

  const size_t NB = (size_t)BATCH * UNITS * sizeof(float);   // 64 MB
  char* ws = (char*)d_ws;
  float* y     = (float*)(ws);                      // RK4 accumulator
  float* vtmp  = (float*)(ws + NB);                 // stage input
  float* z     = (float*)(ws + 2 * NB);             // MLP output
  f16*   h     = (f16*)  (ws + 3 * NB);             // hidden, 4 MB
  f16*   W1t   = (f16*)  (ws + 3 * NB + (size_t)BATCH * HID * sizeof(f16));
  f16*   W2t   = W1t + (size_t)HID * UNITS;
  float* stats = (float*)((char*)(W2t + (size_t)UNITS * HID));

  // one-time (deterministic, cheap) weight convert/transpose + state init
  ode_prep_weights_k<<<(HID * UNITS) / 256, 256, 0, stream>>>(W1, W2, W1t, W2t);
  hipMemcpyAsync(v, x, NB, hipMemcpyDeviceToDevice, stream);

  const float wk[4] = {DT / 6.0f, DT / 3.0f, DT / 3.0f, DT / 6.0f};
  const float ck[4] = {0.5f * DT, 0.5f * DT, DT, 0.0f};

  for (int step = 0; step < NSTEP; ++step) {
    hipMemcpyAsync(y, v, NB, hipMemcpyDeviceToDevice, stream);   // y = v
    const float* vin = v;
    for (int s = 0; s < 4; ++s) {
      ode_gemm1_relu_k<<<BATCH / 64, 128, 0, stream>>>(vin, W1t, b1, h);
      ode_gemm2_relu_k<<<dim3(BATCH / 64, UNITS / 128), 128, 0, stream>>>(h, W2t, b2, z);
      ode_rowstats_k<<<BATCH / 8, 256, 0, stream>>>(vin, z, stats);
      ode_apply_update_k<<<(BATCH * UNITS / 4) / 256, 256, 0, stream>>>(
          z, stats, v, y, vtmp, wk[s], ck[s], (s < 3) ? 1 : 0);
      vin = vtmp;
    }
    hipMemcpyAsync(v, y, NB, hipMemcpyDeviceToDevice, stream);   // v = y
  }
}
